// CausalSelfAttention_58085137711119
// MI455X (gfx1250) — matrix operand order repaired and checked by reading, not compiled, not run on hardware
//
#include <hip/hip_runtime.h>

typedef __bf16 bf16_t;
typedef __attribute__((ext_vector_type(16))) __bf16 v16bf;
typedef __attribute__((ext_vector_type(8)))  float  v8f;
typedef __attribute__((ext_vector_type(4)))  unsigned int u32x4;

union FragBF { v16bf v; u32x4 u[2]; };
union Vec8BF { u32x4 u; __bf16 h[8]; };

// ---- CDNA5 async global->LDS staging (ASYNCcnt path), with safe fallback ----
#if __has_builtin(__builtin_amdgcn_global_load_async_to_lds_b128) && \
    __has_builtin(__builtin_amdgcn_s_wait_asynccnt)
#define HAVE_ASYNC_LDS 1
#else
#define HAVE_ASYNC_LDS 0
#endif

#if HAVE_ASYNC_LDS
// builtin signature (from hipcc diagnostic): (v4i __device__*, v4i __shared__*, Ii, Ii)
typedef int i32x4v __attribute__((vector_size(16)));
#define ASG __attribute__((address_space(1)))
#define ASL __attribute__((address_space(3)))
#endif

__device__ __forceinline__ void cp_g2lds_b128(bf16_t* lds, const bf16_t* g) {
#if HAVE_ASYNC_LDS
  __builtin_amdgcn_global_load_async_to_lds_b128(
      (ASG i32x4v*)g, (ASL i32x4v*)lds, 0, 0);
#else
  *(u32x4*)lds = *(const u32x4*)g;
#endif
}

__device__ __forceinline__ void wait_async_lds() {
#if HAVE_ASYNC_LDS
  __builtin_amdgcn_s_wait_asynccnt(0);
#endif
}

// ---------------- fp32 -> bf16 conversion ----------------
__global__ void f32_to_bf16_kernel(const float* __restrict__ in,
                                   bf16_t* __restrict__ out, int n) {
  int i = blockIdx.x * blockDim.x + threadIdx.x;
  int stride = gridDim.x * blockDim.x;
  for (; i < n; i += stride) out[i] = (bf16_t)in[i];
}

// ---------------- tiled WMMA GEMM: C = A(bf16)[MxK] * B(bf16)[KxN] + bias ----------------
// BM=128 BN=128 BK=32, 256 threads = 8 waves; wave grid 4(rows)x2(cols),
// each wave: 2x4 tiles of 16x16 -> 8 v_wmma per K step.
template <bool OUT_F32>
__global__ __launch_bounds__(256) void gemm_bf16_kernel(
    const bf16_t* __restrict__ A, const bf16_t* __restrict__ Bm,
    const float* __restrict__ bias, void* __restrict__ Cout,
    int M, int N, int K) {
  __shared__ bf16_t As[128 * 32];   // [row][k], row-major, 16B-aligned rows
  __shared__ bf16_t Bt[128 * 32];   // [n][k] (transposed on load)

  const int t = threadIdx.x;
  const int warp = t >> 5, lane = t & 31;
  const int lane16 = lane & 15, hi = lane >> 4;
  const int wrow = (warp >> 1) * 32;   // wave row offset inside block tile
  const int wcol = (warp & 1) * 64;    // wave col offset inside block tile
  const int rowBase = blockIdx.y * 128;
  const int colBase = blockIdx.x * 128;

  const v8f vzero = {0.f, 0.f, 0.f, 0.f, 0.f, 0.f, 0.f, 0.f};
  v8f acc[2][4];
#pragma unroll
  for (int i = 0; i < 2; ++i)
#pragma unroll
    for (int j = 0; j < 4; ++j) acc[i][j] = vzero;

  const int arow = t >> 1, achunk = (t & 1) * 16;   // A: 128 rows x 32 k, 32B/thread
  const int brow = t >> 3, bcol = (t & 7) * 16;     // B: 32 k x 128 n, 32B/thread

  for (int k0 = 0; k0 < K; k0 += 32) {
    __syncthreads();
    // ---- stage A tile (row-major): async global->LDS, no VGPR round-trip ----
    const bf16_t* ag = A + (size_t)(rowBase + arow) * K + k0 + achunk;
    cp_g2lds_b128(&As[arow * 32 + achunk], ag);
    cp_g2lds_b128(&As[arow * 32 + achunk + 8], ag + 8);
    // ---- stage B tile transposed: Bt[n][k] (must pass through VGPRs) ----
    const bf16_t* bg = Bm + (size_t)(k0 + brow) * N + colBase + bcol;
    Vec8BF b0, b1;
    b0.u = *(const u32x4*)bg;
    b1.u = *(const u32x4*)(bg + 8);
#pragma unroll
    for (int j = 0; j < 8; ++j) {
      Bt[(bcol + j) * 32 + brow]     = b0.h[j];
      Bt[(bcol + 8 + j) * 32 + brow] = b1.h[j];
    }
    if (k0 + 32 < K) {  // warm next tiles -> global_prefetch_b8
      __builtin_prefetch(ag + 32, 0, 1);
      __builtin_prefetch(bg + (size_t)32 * N, 0, 1);
    }
    wait_async_lds();
    __syncthreads();

    // ---- fragments per documented gfx1250 16-bit layouts ----
    FragBF afr[2], bfr[4];
#pragma unroll
    for (int i = 0; i < 2; ++i) {
      int m = wrow + i * 16 + lane16;
      afr[i].u[0] = *(const u32x4*)&As[m * 32 + hi * 8];        // K 0..7 / 8..15
      afr[i].u[1] = *(const u32x4*)&As[m * 32 + 16 + hi * 8];   // K 16..23 / 24..31
    }
#pragma unroll
    for (int j = 0; j < 4; ++j) {
      int n = wcol + j * 16 + lane16;
      bfr[j].u[0] = *(const u32x4*)&Bt[n * 32 + hi * 8];        // K 0..7 / 8..15
      bfr[j].u[1] = *(const u32x4*)&Bt[n * 32 + 16 + hi * 8];   // K 16..23 / 24..31
    }
#pragma unroll
    for (int i = 0; i < 2; ++i)
#pragma unroll
      for (int j = 0; j < 4; ++j)
        acc[i][j] = __builtin_amdgcn_wmma_f32_16x16x32_bf16(
            false, afr[i].v, false, bfr[j].v, (short)0, acc[i][j], false, false);
  }

  // ---- epilogue: bias + store (C/D layout: m = r + 8*hi, n = lane16) ----
#pragma unroll
  for (int i = 0; i < 2; ++i) {
#pragma unroll
    for (int j = 0; j < 4; ++j) {
      int n = colBase + wcol + j * 16 + lane16;
      float bv = bias[n];
#pragma unroll
      for (int r = 0; r < 8; ++r) {
        int m = rowBase + wrow + i * 16 + r + hi * 8;
        float v = acc[i][j][r] + bv;
        if constexpr (OUT_F32)
          ((float*)Cout)[(size_t)m * N + n] = v;
        else
          ((bf16_t*)Cout)[(size_t)m * N + n] = (bf16_t)v;
      }
    }
  }
}

// ---------------- flash causal attention ----------------
// Grid: (T/64, H, B). Block: 128 threads = 4 waves, each wave owns 16 query rows.
// qkv: bf16 [B*T, 3072], q at col h*64, k at 1024+h*64, v at 2048+h*64.
// y  : bf16 [B*T, 1024], col h*64+d.
__global__ __launch_bounds__(128) void attn_kernel(const bf16_t* __restrict__ qkv,
                                                   bf16_t* __restrict__ yb) {
  constexpr int T = 2048, C3 = 3072, C = 1024, DH = 64;
  const int qt = blockIdx.x, h = blockIdx.y, b = blockIdx.z;
  const int qbase = qt * 64;

  __shared__ bf16_t Qs[64 * 64];        // [q][d] row-major
  __shared__ bf16_t Ks[64 * 64];        // [kpos][d] row-major (== B layout for Q*K^T)
  __shared__ bf16_t Vt[64 * 64];        // [d][kpos] (transposed: B layout for P*V)
  __shared__ bf16_t Ps[4 * 16 * 64];    // per-wave P tile [16][64]

  const int t = threadIdx.x;
  const int warp = t >> 5, lane = t & 31;
  const int lane16 = lane & 15, hi = lane >> 4;

  // ---- load Q block (row-major -> async direct to LDS) ----
  const bf16_t* qptr = qkv + (size_t)(b * T + qbase) * C3 + h * DH;
#pragma unroll
  for (int i = 0; i < 4; ++i) {
    int idx = t + i * 128;           // 512 x 16B chunks total
    int row = idx >> 3, part = idx & 7;
    cp_g2lds_b128(&Qs[row * 64 + part * 8], qptr + (size_t)row * C3 + part * 8);
  }
  wait_async_lds();
  __syncthreads();

  // ---- Q fragments (A-matrix 16x32, two Dh chunks), kept in registers ----
  FragBF qf[2];
  {
    int row = warp * 16 + lane16;
#pragma unroll
    for (int c = 0; c < 2; ++c) {
      qf[c].u[0] = *(const u32x4*)&Qs[row * 64 + c * 32 + hi * 8];
      qf[c].u[1] = *(const u32x4*)&Qs[row * 64 + c * 32 + 16 + hi * 8];
    }
  }

  const v8f vzero = {0.f, 0.f, 0.f, 0.f, 0.f, 0.f, 0.f, 0.f};
  v8f Oacc[4];
  float mrow[8], lrow[8];
#pragma unroll
  for (int tn = 0; tn < 4; ++tn) Oacc[tn] = vzero;
#pragma unroll
  for (int r = 0; r < 8; ++r) { mrow[r] = -3.0e38f; lrow[r] = 0.f; }

  const bf16_t* kbp = qkv + (size_t)b * T * C3 + C + h * DH;
  const bf16_t* vbp = qkv + (size_t)b * T * C3 + 2 * C + h * DH;
  bf16_t* pw = Ps + warp * 16 * 64;

  for (int kb = 0; kb <= qt; ++kb) {
    __syncthreads();
    // ---- stage K (row-major, async) and V (transposed via VGPRs) ----
#pragma unroll
    for (int i = 0; i < 4; ++i) {
      int idx = t + i * 128;
      int row = idx >> 3, part = idx & 7;
      const size_t roff = (size_t)(kb * 64 + row) * C3 + part * 8;
      cp_g2lds_b128(&Ks[row * 64 + part * 8], kbp + roff);
      Vec8BF vv; vv.u = *(const u32x4*)(vbp + roff);
#pragma unroll
      for (int j = 0; j < 8; ++j) Vt[(part * 8 + j) * 64 + row] = vv.h[j];
    }
    wait_async_lds();
    __syncthreads();

    // ---- S = Q * K^T : 4 n-tiles x 2 K-chunks ----
    v8f Sacc[4];
#pragma unroll
    for (int tn = 0; tn < 4; ++tn) Sacc[tn] = vzero;
#pragma unroll
    for (int tn = 0; tn < 4; ++tn) {
#pragma unroll
      for (int c = 0; c < 2; ++c) {
        FragBF kf;
        int n = tn * 16 + lane16;
        int db = c * 32 + hi * 8;
        kf.u[0] = *(const u32x4*)&Ks[n * 64 + db];
        kf.u[1] = *(const u32x4*)&Ks[n * 64 + db + 16];
        Sacc[tn] = __builtin_amdgcn_wmma_f32_16x16x32_bf16(
            false, qf[c].v, false, kf.v, (short)0, Sacc[tn], false, false);
      }
    }

    // ---- online softmax (rows m = r + 8*hi, cols n = tn*16 + lane16) ----
    const bool diag = (kb == qt);
    float mnew[8];
#pragma unroll
    for (int r = 0; r < 8; ++r) mnew[r] = mrow[r];
#pragma unroll
    for (int tn = 0; tn < 4; ++tn) {
#pragma unroll
      for (int r = 0; r < 8; ++r) {
        float s = Sacc[tn][r] * 0.125f;   // 1/sqrt(64)
        if (diag) {
          int qloc = warp * 16 + r + hi * 8;
          int kloc = tn * 16 + lane16;
          if (kloc > qloc) s = -3.0e38f;
        }
        Sacc[tn][r] = s;
        mnew[r] = fmaxf(mnew[r], s);
      }
    }
#pragma unroll
    for (int mk = 1; mk <= 8; mk <<= 1)
#pragma unroll
      for (int r = 0; r < 8; ++r)
        mnew[r] = fmaxf(mnew[r], __shfl_xor(mnew[r], mk, 32));

    float rsum[8];
#pragma unroll
    for (int r = 0; r < 8; ++r) rsum[r] = 0.f;
#pragma unroll
    for (int tn = 0; tn < 4; ++tn) {
#pragma unroll
      for (int r = 0; r < 8; ++r) {
        float p = __expf(Sacc[tn][r] - mnew[r]);
        Sacc[tn][r] = p;
        rsum[r] += p;
      }
    }
#pragma unroll
    for (int mk = 1; mk <= 8; mk <<= 1)
#pragma unroll
      for (int r = 0; r < 8; ++r) rsum[r] += __shfl_xor(rsum[r], mk, 32);

#pragma unroll
    for (int r = 0; r < 8; ++r) {
      float alpha = __expf(mrow[r] - mnew[r]);
      lrow[r] = lrow[r] * alpha + rsum[r];
      mrow[r] = mnew[r];
#pragma unroll
      for (int tn = 0; tn < 4; ++tn) Oacc[tn][r] *= alpha;
    }

    // ---- P -> LDS (per-wave region; LDS in-order within wave) ----
#pragma unroll
    for (int tn = 0; tn < 4; ++tn)
#pragma unroll
      for (int r = 0; r < 8; ++r)
        pw[(r + hi * 8) * 64 + tn * 16 + lane16] = (bf16_t)Sacc[tn][r];

    // ---- O += P * V : 4 d-tiles x 2 kpos-chunks ----
#pragma unroll
    for (int tn = 0; tn < 4; ++tn) {
#pragma unroll
      for (int c = 0; c < 2; ++c) {
        FragBF pf, vf;
        pf.u[0] = *(const u32x4*)&pw[lane16 * 64 + c * 32 + hi * 8];
        pf.u[1] = *(const u32x4*)&pw[lane16 * 64 + c * 32 + 16 + hi * 8];
        int nd = tn * 16 + lane16;
        int kbse = c * 32 + hi * 8;
        vf.u[0] = *(const u32x4*)&Vt[nd * 64 + kbse];
        vf.u[1] = *(const u32x4*)&Vt[nd * 64 + kbse + 16];
        Oacc[tn] = __builtin_amdgcn_wmma_f32_16x16x32_bf16(
            false, pf.v, false, vf.v, (short)0, Oacc[tn], false, false);
      }
    }
  }

  // ---- normalize and store y (bf16) ----
  bf16_t* yp = yb + (size_t)(b * T) * C + h * DH;
#pragma unroll
  for (int tn = 0; tn < 4; ++tn) {
#pragma unroll
    for (int r = 0; r < 8; ++r) {
      int q = qbase + warp * 16 + r + hi * 8;
      int d = tn * 16 + lane16;
      yp[(size_t)q * C + d] = (bf16_t)(Oacc[tn][r] / lrow[r]);
    }
  }
}

// ---------------- launch ----------------
extern "C" void kernel_launch(void* const* d_in, const int* in_sizes, int n_in,
                              void* d_out, int out_size, void* d_ws, size_t ws_size,
                              hipStream_t stream) {
  const float* x      = (const float*)d_in[0];
  const float* W_attn = (const float*)d_in[1];
  const float* b_attn = (const float*)d_in[2];
  const float* W_proj = (const float*)d_in[3];
  const float* b_proj = (const float*)d_in[4];

  // workspace layout (bytes)
  char* ws = (char*)d_ws;
  bf16_t* XB  = (bf16_t*)(ws);                 // 4096x1024 bf16 :  8 MB
  bf16_t* WAB = (bf16_t*)(ws + 8388608);       // 1024x3072 bf16 :  6 MB
  bf16_t* WPB = (bf16_t*)(ws + 14680064);      // 1024x1024 bf16 :  2 MB
  bf16_t* QKV = (bf16_t*)(ws + 16777216);      // 4096x3072 bf16 : 24 MB
  bf16_t* YB  = (bf16_t*)(ws + 41943040);      // 4096x1024 bf16 :  8 MB

  f32_to_bf16_kernel<<<2048, 256, 0, stream>>>(x,      XB,  4194304);
  f32_to_bf16_kernel<<<2048, 256, 0, stream>>>(W_attn, WAB, 3145728);
  f32_to_bf16_kernel<<<1024, 256, 0, stream>>>(W_proj, WPB, 1048576);

  // QKV = x @ W_attn + b_attn  -> bf16 [4096, 3072]
  gemm_bf16_kernel<false><<<dim3(24, 32), 256, 0, stream>>>(
      XB, WAB, b_attn, (void*)QKV, 4096, 3072, 1024);

  // flash causal attention -> YB bf16 [4096, 1024]
  attn_kernel<<<dim3(32, 16, 2), 128, 0, stream>>>(QKV, YB);

  // out = y @ W_proj + b_proj -> f32
  gemm_bf16_kernel<true><<<dim3(8, 32), 256, 0, stream>>>(
      YB, WPB, b_proj, d_out, 4096, 1024, 1024);
}
